// convDU_42417097016952
// MI455X (gfx1250) — matrix operand order repaired and checked by reading, not compiled, not run on hardware
//
#include <hip/hip_runtime.h>

// Problem constants (from the reference)
#define C_DIM 2048
#define H_DIM 48
#define W_DIM 96
#define N_DIM 4
#define JTOT  (N_DIM * W_DIM)     // 384 GEMM columns (n,w)
#define HW    (H_DIM * W_DIM)     // 4608
#define CHW   ((size_t)C_DIM * HW)

// Workgroup GEMM tiling
#define BM 128                    // workgroup M tile (4 waves of 32)
#define BJ 64                     // workgroup J tile (2 waves of 32)
#define BK 32                     // K step (WMMA K)
#define LDSS 40                   // LDS row stride in bf16 (80B = 64B data + 16B pad)

typedef __attribute__((ext_vector_type(16))) __bf16 bfx16;
typedef __attribute__((ext_vector_type(8)))  __bf16 bfx8;
typedef __attribute__((ext_vector_type(8)))  float  fx8;
typedef __attribute__((ext_vector_type(4)))  unsigned int u32x4;
typedef __attribute__((ext_vector_type(4)))  int i32x4;
typedef __attribute__((ext_vector_type(8)))  int i32x8;

union ABFrag { bfx16 v; bfx8 h[2]; };

#if __has_builtin(__builtin_amdgcn_tensor_load_to_lds) && __has_builtin(__builtin_amdgcn_s_wait_tensorcnt)
#define HAVE_TDM 1
#else
#define HAVE_TDM 0
#endif

#if HAVE_TDM
// 2-D TDM load: tile (tile_rows x 32) bf16 elements from a row-major tensor with
// row stride 2048 elements into LDS, padding LDS rows 64B data + 16B (-> 80B stride).
__device__ __forceinline__ void tdm_load_2d(unsigned lds_addr, const void* gaddr,
                                            unsigned tensor_rows, unsigned tile_rows) {
    unsigned long long ga = (unsigned long long)gaddr;
    u32x4 g0 = { 1u,                                   // count=1, no gather
                 lds_addr,                             // LDS byte address
                 (unsigned)ga,                         // global_addr[31:0]
                 (unsigned)((ga >> 32) & 0x01FFFFFFu)  // global_addr[56:32]
                     | 0x80000000u };                  // type=2 ("image")
    i32x8 g1 = { (int)( (1u << 16)                     // data_size = 2 bytes
                      | (1u << 20)                     // pad_enable
                      | (3u << 22)                     // pad_interval: 16 DWORDs (64B)
                      | (3u << 25) ),                  // pad_amount: 4 DWORDs (16B)
                 (int)(2048u << 16),                   // tensor_dim0 = 2048 (lo<<16)
                 (int)(tensor_rows << 16),             // dim0 hi=0 | tensor_dim1 lo
                 (int)(32u << 16),                     // dim1 hi=0 | tile_dim0 = 32
                 (int)tile_rows,                       // tile_dim1 | tile_dim2=0
                 2048,                                 // tensor_dim0_stride lo
                 0, 0 };                               // stride hi, dim1_stride
    i32x4 z4 = {0, 0, 0, 0};
#if defined(__clang_major__) && __clang_major__ >= 23
    i32x8 z8 = {0, 0, 0, 0, 0, 0, 0, 0};
    __builtin_amdgcn_tensor_load_to_lds(g0, g1, z4, z4, z8, 0);
#else
    __builtin_amdgcn_tensor_load_to_lds(g0, g1, z4, z4, 0);
#endif
}
#endif

// --- one-time: extract center tap of W and convert to bf16, layout [o][c] ---
__global__ void convert_w_kernel(const float* __restrict__ Wfull,
                                 __bf16* __restrict__ Wbf) {
    int idx = blockIdx.x * blockDim.x + threadIdx.x;   // idx = o*2048 + c
    int o = idx >> 11;
    int c = idx & 2047;
    float v = Wfull[(size_t)(o * C_DIM + c) * 9 + 4];  // W[o, c, 4, 0]
    Wbf[idx] = (__bf16)v;
}

// --- one-time: row h=0 is a pure copy; also seed transposed bf16 carry ---
__global__ void init_row0_kernel(const float* __restrict__ fea,
                                 float* __restrict__ out,
                                 __bf16* __restrict__ Xt) {
    int idx = blockIdx.x * blockDim.x + threadIdx.x;   // idx = j*2048 + o
    int j = idx >> 11;
    int o = idx & 2047;
    int n = j / W_DIM, w = j - n * W_DIM;
    size_t off = (size_t)n * CHW + (size_t)o * HW + w; // h = 0
    float v = fea[off];
    out[off] = v;
    Xt[idx] = (__bf16)v;                               // Xt[j][o], c contiguous
}

// --- one scan step: Y = relu(Wm@X + b) + resid ---
// 8 waves / block: workgroup tile 128(M) x 64(J); TDM DMAs the A/B panels into
// double-buffered LDS while all waves compute from the other buffer.
__global__ __launch_bounds__(256)
void gemm_step_kernel(const __bf16* __restrict__ Wbf,   // [2048][2048] bf16
                      const __bf16* __restrict__ Xin,   // [384][2048]  bf16 (transposed carry)
                      const float*  __restrict__ bias,  // [2048]
                      const float*  resid,              // NCHW f32 (may alias out)
                      float*        out,                // NCHW f32
                      __bf16*       __restrict__ Xout,  // [384][2048] bf16
                      int h) {
    __shared__ __bf16 ldsA[2][BM * LDSS];   // 2 x 10240 B
    __shared__ __bf16 ldsB[2][BJ * LDSS];   // 2 x  5120 B

    const int tid   = threadIdx.x;
    const int lane  = tid & 31;
    const int wave  = tid >> 5;            // 0..7
    const int laneN = lane & 15;           // row (A) / col (B) within 16x16 tile
    const int hl    = lane >> 4;           // which K-half this lane carries
    const int waveM = wave >> 1;           // 0..3
    const int waveJ = wave & 1;            // 0..1
    const int mWG   = blockIdx.x * BM;
    const int jWG   = blockIdx.y * BJ;
    const int mBase = mWG + waveM * 32;
    const int jBase = jWG + waveJ * 32;

    // accumulators seeded with bias[o]; D-layout row = r + 8*hl
    fx8 acc[2][2];
#pragma unroll
    for (int mi = 0; mi < 2; ++mi) {
        fx8 bv = *(const fx8*)(bias + mBase + mi * 16 + hl * 8);
        acc[mi][0] = bv;
        acc[mi][1] = bv;
    }

    // fragment base offsets inside the LDS panels (elements)
    const int ra0 = (waveM * 32 + laneN) * LDSS;
    const int ra1 = ra0 + 16 * LDSS;
    const int rb0 = (waveJ * 32 + laneN) * LDSS;
    const int rb1 = rb0 + 16 * LDSS;

#if HAVE_TDM
    const char* gAbase = (const char*)(Wbf + (size_t)mWG * C_DIM);
    const char* gBbase = (const char*)(Xin + (size_t)jWG * C_DIM);
    const unsigned ldsAaddr[2] = { (unsigned)(size_t)&ldsA[0][0],
                                   (unsigned)(size_t)&ldsA[1][0] };
    const unsigned ldsBaddr[2] = { (unsigned)(size_t)&ldsB[0][0],
                                   (unsigned)(size_t)&ldsB[1][0] };
    if (wave == 0) {                         // prime buffer 0 with k0 = 0
        tdm_load_2d(ldsAaddr[0], gAbase, C_DIM, BM);
        tdm_load_2d(ldsBaddr[0], gBbase, JTOT, BJ);
        __builtin_amdgcn_s_wait_tensorcnt((short)0);
    }
    __syncthreads();
#else
    // fallback: register staging (A 32B/thread, B 16B/thread)
    const int sArow = tid >> 1;
    const int sAcol = (tid & 1) * 16;
    const int sBrow = tid >> 2;
    const int sBcol = (tid & 3) * 8;
    const __bf16* gA = Wbf + (size_t)(mWG + sArow) * C_DIM + sAcol;
    const __bf16* gB = Xin + (size_t)(jWG + sBrow) * C_DIM + sBcol;
    const int lAoff = sArow * LDSS + sAcol;
    const int lBoff = sBrow * LDSS + sBcol;
    {
        bfx16 aS = *(const bfx16*)(gA);
        bfx8  bS = *(const bfx8*)(gB);
        *(bfx16*)(&ldsA[0][lAoff]) = aS;
        *(bfx8*)(&ldsB[0][lBoff])  = bS;
    }
    __syncthreads();
#endif

    int cur = 0;
    for (int k0 = 0; k0 < C_DIM; k0 += BK) {
        const bool hasNext = (k0 + BK) < C_DIM;

#if HAVE_TDM
        if (hasNext && wave == 0) {          // DMA next panels while we compute
            tdm_load_2d(ldsAaddr[cur ^ 1], gAbase + (size_t)(k0 + BK) * 2, C_DIM, BM);
            tdm_load_2d(ldsBaddr[cur ^ 1], gBbase + (size_t)(k0 + BK) * 2, JTOT, BJ);
        }
#else
        bfx16 aN; bfx8 bN;
        if (hasNext) {
            aN = *(const bfx16*)(gA + k0 + BK);
            bN = *(const bfx8*)(gB + k0 + BK);
        }
#endif

        const __bf16* As = &ldsA[cur][0];
        const __bf16* Bs = &ldsB[cur][0];
        ABFrag a0, a1, b0, b1;
        // A 16x32 bf16: lane m holds K = hl*8..+8 (V0-3) and 16+hl*8..+8 (V4-7)
        a0.h[0] = *(const bfx8*)(As + ra0 + hl * 8);
        a0.h[1] = *(const bfx8*)(As + ra0 + 16 + hl * 8);
        a1.h[0] = *(const bfx8*)(As + ra1 + hl * 8);
        a1.h[1] = *(const bfx8*)(As + ra1 + 16 + hl * 8);
        // B 32x16 bf16: lane n holds K = hl*8..+8 (V0-3) and 16+hl*8..+8 (V4-7)
        b0.h[0] = *(const bfx8*)(Bs + rb0 + hl * 8);
        b0.h[1] = *(const bfx8*)(Bs + rb0 + 16 + hl * 8);
        b1.h[0] = *(const bfx8*)(Bs + rb1 + hl * 8);
        b1.h[1] = *(const bfx8*)(Bs + rb1 + 16 + hl * 8);

        acc[0][0] = __builtin_amdgcn_wmma_f32_16x16x32_bf16(false, a0.v, false, b0.v, (short)0, acc[0][0], false, false);
        acc[0][1] = __builtin_amdgcn_wmma_f32_16x16x32_bf16(false, a0.v, false, b1.v, (short)0, acc[0][1], false, false);
        acc[1][0] = __builtin_amdgcn_wmma_f32_16x16x32_bf16(false, a1.v, false, b0.v, (short)0, acc[1][0], false, false);
        acc[1][1] = __builtin_amdgcn_wmma_f32_16x16x32_bf16(false, a1.v, false, b1.v, (short)0, acc[1][1], false, false);

#if HAVE_TDM
        if (hasNext && wave == 0)
            __builtin_amdgcn_s_wait_tensorcnt((short)0);   // next buffer landed
#else
        if (hasNext) {
            *(bfx16*)(&ldsA[cur ^ 1][lAoff]) = aN;
            *(bfx8*)(&ldsB[cur ^ 1][lBoff])  = bN;
        }
#endif
        __syncthreads();                     // one barrier per K-step
        cur ^= 1;
    }

    // Epilogue: relu + residual, f32 out (NCHW), bf16 packed store of next carry
#pragma unroll
    for (int mi = 0; mi < 2; ++mi) {
#pragma unroll
        for (int ji = 0; ji < 2; ++ji) {
            int j = jBase + ji * 16 + laneN;
            int n = j / W_DIM;
            int w = j - n * W_DIM;
            int oBase = mBase + mi * 16 + hl * 8;      // 8 consecutive channels
            size_t off = (size_t)n * CHW + (size_t)oBase * HW + (size_t)h * W_DIM + w;
            bfx8 pack;
#pragma unroll
            for (int r = 0; r < 8; ++r) {
                float v = acc[mi][ji][r];
                float res = resid[off + (size_t)r * HW];   // read-before-write per element
                v = fmaxf(v, 0.f) + res;
                out[off + (size_t)r * HW] = v;
                pack[r] = (__bf16)v;
            }
            *(bfx8*)(Xout + (size_t)j * C_DIM + oBase) = pack;  // 16B contiguous store
        }
    }
}

extern "C" void kernel_launch(void* const* d_in, const int* in_sizes, int n_in,
                              void* d_out, int out_size, void* d_ws, size_t ws_size,
                              hipStream_t stream) {
    const float* fea   = (const float*)d_in[0];
    const float* Wfull = (const float*)d_in[1];
    const float* bias  = (const float*)d_in[2];
    float* out = (float*)d_out;

    char* ws = (char*)d_ws;
    __bf16* Wbf = (__bf16*)ws;                                        // 8 MB
    __bf16* Xt0 = (__bf16*)(ws + (size_t)C_DIM * C_DIM * sizeof(__bf16));
    __bf16* Xt1 = Xt0 + (size_t)JTOT * C_DIM;                         // 1.5 MB each

    convert_w_kernel<<<(C_DIM * C_DIM) / 256, 256, 0, stream>>>(Wfull, Wbf);
    init_row0_kernel<<<(JTOT * C_DIM) / 256, 256, 0, stream>>>(fea, out, Xt0);

    dim3 grid(C_DIM / BM, JTOT / BJ);   // 16 x 6 workgroups, 8 waves each
    __bf16* bufs[2] = {Xt0, Xt1};

    // Down pass: h = 1..47, residual from fea. bufs[0] holds row 0.
    for (int h = 1; h < H_DIM; ++h) {
        gemm_step_kernel<<<grid, 256, 0, stream>>>(Wbf, bufs[(h - 1) & 1], bias,
                                                   fea, out, bufs[h & 1], h);
    }
    // Up pass: carry = down[47] (in bufs[1]); h = 46..0, residual = down[h]
    // read from d_out and overwritten in place.
    for (int i = 0; i < H_DIM - 1; ++i) {
        int h = H_DIM - 2 - i;
        gemm_step_kernel<<<grid, 256, 0, stream>>>(Wbf, bufs[(i + 1) & 1], bias,
                                                   out, out, bufs[i & 1], h);
    }
}
